// Histogram_Matching_63041529971132
// MI455X (gfx1250) — compile-verified, matrix operand order checked by reading, not run on hardware
//
#include <hip/hip_runtime.h>
#include <stdint.h>

// ---------------------------------------------------------------------------
// Histogram matching, MI455X (gfx1250).
//   Pass 1: per-(b,c)-row 256-bin histograms of dst & ref  (bandwidth bound)
//   Pass 2: CDF via WMMA IU8 "ones-matmul" (exact byte-split) + two-pointer LUT
//   Pass 3: LUT gather apply                                 (bandwidth bound)
// Traffic ~400MB -> ~17us at 23.3 TB/s; passes 1/3 use float4 (B128) streams.
// ---------------------------------------------------------------------------

#define HW      (1024 * 1024)
#define BC      24
#define CHUNKS  64                  // blocks per row in streaming passes
#define PIX_PER_BLK   (HW / CHUNKS) // 16384 pixels
#define VEC_PER_BLK   (PIX_PER_BLK / 4)

typedef __attribute__((ext_vector_type(8))) int v8i;

__device__ __forceinline__ int bin256(float x) {
    int b = (int)(x * 256.0f);
    return b < 0 ? 0 : (b > 255 ? 255 : b);
}
__device__ __forceinline__ int bin255(float x) {
    int b = (int)(x * 255.0f);
    return b < 0 ? 0 : (b > 255 ? 255 : b);
}

// ---------------- Pass 1: histograms (dst & ref together) -------------------
__global__ void hm_hist_kernel(const float* __restrict__ dst,
                               const float* __restrict__ ref,
                               unsigned* __restrict__ ghist) { // [2][24][256]
    __shared__ unsigned hd[256], hr[256];
    const int tid = threadIdx.x;
    hd[tid] = 0u; hr[tid] = 0u;
    __syncthreads();

    const int row = blockIdx.y;
    const size_t pix0 = (size_t)row * HW + (size_t)blockIdx.x * PIX_PER_BLK;
    const float4* d4 = (const float4*)(dst + pix0);
    const float4* r4 = (const float4*)(ref + pix0);

    for (int i = tid; i < VEC_PER_BLK; i += 256) {
        __builtin_prefetch(&d4[i + 1024], 0, 0);   // -> global_prefetch_b8
        __builtin_prefetch(&r4[i + 1024], 0, 0);
        float4 a = d4[i];
        float4 b = r4[i];
        atomicAdd(&hd[bin256(a.x)], 1u);  // ds_add_u32 (LDS-privatized)
        atomicAdd(&hd[bin256(a.y)], 1u);
        atomicAdd(&hd[bin256(a.z)], 1u);
        atomicAdd(&hd[bin256(a.w)], 1u);
        atomicAdd(&hr[bin256(b.x)], 1u);
        atomicAdd(&hr[bin256(b.y)], 1u);
        atomicAdd(&hr[bin256(b.z)], 1u);
        atomicAdd(&hr[bin256(b.w)], 1u);
    }
    __syncthreads();
    atomicAdd(&ghist[row * 256 + tid], hd[tid]);
    atomicAdd(&ghist[BC * 256 + row * 256 + tid], hr[tid]);
}

// ---------------- Pass 2: CDFs via WMMA IU8 + two-pointer LUT ---------------
// Counts (<= 2^20) are split into 3 byte planes; each plane is multiplied by
// an all-ones 64x16 u8 B-matrix with i32 accumulation (V_WMMA_I32_16X16X64_IU8)
// to get exact per-64-column prefix "bases" (the reference's triu-bmm). Then a
// 192-thread integer scan finishes the cumsum; float(cum)*2^-20 is bit-exact.
__global__ void hm_cdf_lut_kernel(const unsigned* __restrict__ ghist,
                                  int* __restrict__ gtab,    // [24][256]
                                  int* __restrict__ gbase) { // [2][32][4]
    __shared__ unsigned cnt[2][32][256];  // 64 KB; later holds CDF f32 bits
    const int tid = threadIdx.x;

    // load histograms, zero-pad rows 24..31
    for (int idx = tid; idx < 2 * 32 * 256; idx += 256) {
        int img = idx >> 13;
        int r   = (idx >> 8) & 31;
        int col = idx & 255;
        cnt[img][r][col] = (r < BC) ? ghist[img * (BC * 256) + r * 256 + col] : 0u;
    }
    __syncthreads();

    const int wave = tid >> 5;
    const int lane = tid & 31;
    if (wave < 4) {                 // 4 waves: (img, mtile) in {0,1}x{0,1}
        const int img = wave >> 1;
        const int mt  = wave & 1;
        const int Lm  = lane & 15;
        const int hi  = lane >> 4;
        v8i c0 = {0,0,0,0,0,0,0,0}, c1 = {0,0,0,0,0,0,0,0}, c2 = {0,0,0,0,0,0,0,0};
        const v8i onesB = {0x01010101,0x01010101,0x01010101,0x01010101,
                           0x01010101,0x01010101,0x01010101,0x01010101};
        for (int sb = 0; sb < 4; ++sb) {   // 4 super-blocks of 64 columns
            // emit pre-accumulation bases (C/D layout: lane<16 -> M=r, lane>=16 -> M=r+8)
            if (lane == 0) {
                for (int r = 0; r < 8; ++r)
                    gbase[(img * 32 + mt * 16 + r) * 4 + sb] =
                        c0[r] + (c1[r] << 8) + (c2[r] << 16);
            }
            if (lane == 16) {
                for (int r = 0; r < 8; ++r)
                    gbase[(img * 32 + mt * 16 + 8 + r) * 4 + sb] =
                        c0[r] + (c1[r] << 8) + (c2[r] << 16);
            }
            // A (8-bit 16x64): lane row = Lm; VGPR v holds K = 16*(v>>1)+4*(v&1)+8*hi + j
            v8i a0, a1, a2;
            const int m = mt * 16 + Lm;
            for (int v = 0; v < 8; ++v) {
                const int kbase = 16 * (v >> 1) + 4 * (v & 1) + 8 * hi;
                unsigned p0 = 0, p1 = 0, p2 = 0;
                for (int j = 0; j < 4; ++j) {
                    unsigned k = cnt[img][m][sb * 64 + kbase + j];
                    p0 |= (k & 0xFFu) << (8 * j);
                    p1 |= ((k >> 8) & 0xFFu) << (8 * j);
                    p2 |= ((k >> 16) & 0xFFu) << (8 * j);
                }
                a0[v] = (int)p0; a1[v] = (int)p1; a2[v] = (int)p2;
            }
            c0 = __builtin_amdgcn_wmma_i32_16x16x64_iu8(false, a0, false, onesB, c0, false, false);
            c1 = __builtin_amdgcn_wmma_i32_16x16x64_iu8(false, a1, false, onesB, c1, false, false);
            c2 = __builtin_amdgcn_wmma_i32_16x16x64_iu8(false, a2, false, onesB, c2, false, false);
        }
    }
    __syncthreads();

    // exact integer scan inside each 64-col super-block; store CDF as f32 bits
    if (tid < 192) {                        // 2 imgs * 24 rows * 4 super-blocks
        const int img = tid / 96;
        const int rem = tid % 96;
        const int m   = rem >> 2;
        const int sb  = rem & 3;
        unsigned cum  = (unsigned)gbase[(img * 32 + m) * 4 + sb];
        const float inv = 1.0f / 1048576.0f;      // counts/HW exact in f32
        for (int t = 0; t < 64; ++t) {
            cum += cnt[img][m][sb * 64 + t];
            cnt[img][m][sb * 64 + t] = __float_as_uint((float)cum * inv);
        }
    }
    __syncthreads();

    // faithful two-pointer transfer table (lax.scan semantics), one row/thread
    if (tid < BC) {
        const int m = tid;
        int j = 1;
        gtab[m * 256 + 0] = 0;
        for (int i = 1; i < 256; ++i) {
            const float d = __uint_as_float(cnt[0][m][i]);
            // searchsorted(cdf_ref, d, side='left')
            int lo = 0, hi2 = 256;
            while (lo < hi2) {
                int mid = (lo + hi2) >> 1;
                if (__uint_as_float(cnt[1][m][mid]) < d) lo = mid + 1; else hi2 = mid;
            }
            int cand = j > lo ? j : lo;                     // cand in [1,256]
            float cr = __uint_as_float(cnt[1][m][cand - 1]);
            bool ok = (j < 256) && (cand <= 255) && (d >= cr);
            gtab[m * 256 + i] = ok ? cand : 255;
            j = ok ? cand : 256;
        }
        gtab[m * 256 + 255] = 255;
    }
}

// ---------------- Pass 3: apply LUT (note preserved b*c indexing bug) -------
__global__ void hm_apply_kernel(const float* __restrict__ dst,
                                const int* __restrict__ gtab,
                                float* __restrict__ out) {
    __shared__ float lut[256];
    const int tid = threadIdx.x;
    const int bc  = blockIdx.y;
    const int b   = bc / 3, c = bc % 3;
    const int src = b * c;                     // faithful to tables[b*c]
    lut[tid] = (float)gtab[src * 256 + tid] / 255.0f;
    __syncthreads();

    const size_t pix0 = (size_t)bc * HW + (size_t)blockIdx.x * PIX_PER_BLK;
    const float4* d4 = (const float4*)(dst + pix0);
    float4* o4 = (float4*)(out + pix0);
    for (int i = tid; i < VEC_PER_BLK; i += 256) {
        __builtin_prefetch(&d4[i + 1024], 0, 0);
        float4 a = d4[i];
        float4 o;
        o.x = lut[bin255(a.x)];
        o.y = lut[bin255(a.y)];
        o.z = lut[bin255(a.z)];
        o.w = lut[bin255(a.w)];
        o4[i] = o;
    }
}

// ---------------------------------------------------------------------------
extern "C" void kernel_launch(void* const* d_in, const int* in_sizes, int n_in,
                              void* d_out, int out_size, void* d_ws, size_t ws_size,
                              hipStream_t stream) {
    const float* dst = (const float*)d_in[0];
    const float* ref = (const float*)d_in[1];
    float* out = (float*)d_out;

    uint8_t* ws = (uint8_t*)d_ws;
    unsigned* g_hist = (unsigned*)ws;                       // [2][24][256] u32 = 49152 B
    int*      g_tab  = (int*)(ws + 49152);                  // [24][256]    i32 = 24576 B
    int*      g_base = (int*)(ws + 49152 + 24576);          // [2][32][4]   i32 =  1024 B

    hipMemsetAsync(g_hist, 0, 49152, stream);               // graph-capturable

    dim3 gs(CHUNKS, BC);
    hm_hist_kernel<<<gs, 256, 0, stream>>>(dst, ref, g_hist);
    hm_cdf_lut_kernel<<<1, 256, 0, stream>>>(g_hist, g_tab, g_base);
    hm_apply_kernel<<<gs, 256, 0, stream>>>(dst, g_tab, out);
}
